// LSTMTagger_78099685310775
// MI455X (gfx1250) — matrix operand order repaired and checked by reading, not compiled, not run on hardware
//
#include <hip/hip_runtime.h>
#include <hip/hip_bf16.h>
#include <math.h>

// Problem dims
#define TAGS   9
#define EMBED  300
#define HIDDEN 1000
#define BATCH  32
#define SEQ    200

// Padded dims for WMMA tiling (16x16x32 bf16)
#define KPAD   320              // EMBED padded to multiple of 32
#define HPAD   1024             // HIDDEN padded per gate
#define GPAD   4096             // 4 * HPAD
#define ROWS   (SEQ * BATCH)    // 6400 (multiple of 16)

typedef __attribute__((ext_vector_type(16))) __bf16 v16bf;
typedef __attribute__((ext_vector_type(8)))  float  v8f;
typedef __attribute__((ext_vector_type(4)))  unsigned v4u;

union AFrag { v16bf v; unsigned u[8]; };
union BFrag { v16bf v; v4u q[2]; };

__device__ __forceinline__ unsigned short f32_to_bf16_bits(float f) {
  union { float f; unsigned u; } x; x.f = f;
  unsigned r = x.u + 0x7FFFu + ((x.u >> 16) & 1u);   // round-to-nearest-even
  return (unsigned short)(r >> 16);
}

// A fragment (16x32 bf16, row-major source, lda in elements).
// Lane L (m = L&15, hi = L>>4): VGPR v holds bf16 pair at k = (v<4 ? 2v : 8+2v) + 8*hi
__device__ __forceinline__ v16bf load_a(const unsigned short* __restrict__ p, int lda) {
  const int lane = threadIdx.x & 31;
  const int m = lane & 15, hi = lane >> 4;
  const unsigned short* row = p + (size_t)m * lda;
  AFrag f;
#pragma unroll
  for (int v = 0; v < 8; ++v) {
    int k = ((v < 4) ? (2 * v) : (8 + 2 * v)) + hi * 8;
    f.u[v] = *(const unsigned*)(row + k);
  }
  return f.v;
}

// B fragment (32x16 bf16, K-major source: B[k][n], ldb in elements).
// Lane n = L&15, hi = L>>4; element s holds k = (s<8 ? s : 8+s) + 8*hi.
__device__ __forceinline__ v16bf load_b(const unsigned short* __restrict__ p, int ldb) {
  const int lane = threadIdx.x & 31;
  const int n = lane & 15, hi = lane >> 4;
  union { v16bf v; unsigned short s[16]; } f;
#pragma unroll
  for (int s = 0; s < 16; ++s) {
    int k = ((s < 8) ? s : (8 + s)) + hi * 8;
    f.s[s] = p[(size_t)k * ldb + n];
  }
  return f.v;
}

// B fragment from LDS tile [32 k][16 n]: same k order as load_b
__device__ __forceinline__ v16bf load_b_lds(const unsigned short* p) {
  return load_b(p, 16);
}

__device__ __forceinline__ v8f wmma_bf16(v16bf a, v16bf b, v8f c) {
  return __builtin_amdgcn_wmma_f32_16x16x32_bf16(false, a, false, b, (short)0, c, false, false);
}

__device__ __forceinline__ v8f vzero8() { v8f z = {0.f,0.f,0.f,0.f,0.f,0.f,0.f,0.f}; return z; }

// ---------------- init / layout kernels ----------------

__global__ void init_misc(const float* __restrict__ b_ih, const float* __restrict__ b_hh,
                          float* __restrict__ bias, unsigned short* __restrict__ h_buf,
                          unsigned* __restrict__ bar) {
  size_t i = (size_t)blockIdx.x * blockDim.x + threadIdx.x;
  if (i == 0) *bar = 0u;                        // reset grid barrier every launch (replay-safe)
  if (i < GPAD) {
    int g = (int)(i >> 10), c = (int)(i & 1023);
    bias[i] = (c < HIDDEN) ? (b_ih[g * HIDDEN + c] + b_hh[g * HIDDEN + c]) : 0.f;
  }
  if (i < (size_t)2 * BATCH * HPAD) h_buf[i] = 0;  // zero both h ping-pong buffers
}

// W_ih [4000,300] -> Bih [KPAD][GPAD] K-major bf16, per-gate padded columns
__global__ void conv_wih(const float* __restrict__ W, unsigned short* __restrict__ B) {
  size_t i = (size_t)blockIdx.x * blockDim.x + threadIdx.x;
  if (i >= (size_t)KPAD * GPAD) return;
  int k = (int)(i / GPAD), n = (int)(i % GPAD);
  int g = n >> 10, c = n & 1023;
  float v = (k < EMBED && c < HIDDEN) ? W[(size_t)(g * HIDDEN + c) * EMBED + k] : 0.f;
  B[i] = f32_to_bf16_bits(v);
}

// W_hh [4000,1000] -> Bhh [HPAD][GPAD] K-major bf16
__global__ void conv_whh(const float* __restrict__ W, unsigned short* __restrict__ B) {
  size_t i = (size_t)blockIdx.x * blockDim.x + threadIdx.x;
  if (i >= (size_t)HPAD * GPAD) return;
  int k = (int)(i / GPAD), n = (int)(i % GPAD);
  int g = n >> 10, c = n & 1023;
  float v = (k < HIDDEN && c < HIDDEN) ? W[(size_t)(g * HIDDEN + c) * HIDDEN + k] : 0.f;
  B[i] = f32_to_bf16_bits(v);
}

// gather embeddings -> emb [ROWS][KPAD] bf16 (row r = t*BATCH + b)
__global__ void embed_gather(const int* __restrict__ x, const float* __restrict__ ew,
                             unsigned short* __restrict__ emb) {
  size_t i = (size_t)blockIdx.x * blockDim.x + threadIdx.x;
  if (i >= (size_t)ROWS * KPAD) return;
  int r = (int)(i / KPAD), e = (int)(i % KPAD);
  float v = (e < EMBED) ? ew[(size_t)x[r] * EMBED + e] : 0.f;
  emb[i] = f32_to_bf16_bits(v);
}

// ---------------- xg = emb @ W_ih^T + (b_ih+b_hh) : [ROWS][GPAD] f32 ----------------
// grid (64, 400), block 128 (4 waves): wave -> one 16x16 tile, K = 320
__global__ void __launch_bounds__(128) xg_gemm(const unsigned short* __restrict__ emb,
                                               const unsigned short* __restrict__ Bih,
                                               const float* __restrict__ bias,
                                               float* __restrict__ xg) {
  const int wave = threadIdx.x >> 5;
  const int lane = threadIdx.x & 31;
  const int ntile = blockIdx.x * 4 + wave;   // 0..255
  const int mtile = blockIdx.y;              // 0..399
  v8f acc = vzero8();
  const unsigned short* aBase = emb + (size_t)mtile * 16 * KPAD;
#pragma unroll 2
  for (int kb = 0; kb < KPAD / 32; ++kb) {
    v16bf a = load_a(aBase + kb * 32, KPAD);
    v16bf b = load_b(Bih + (size_t)(kb * 32) * GPAD + ntile * 16, GPAD);
    acc = wmma_bf16(a, b, acc);
  }
  const int col = lane & 15, hi = lane >> 4;
  const float bv = bias[ntile * 16 + col];
#pragma unroll
  for (int r = 0; r < 8; ++r) {
    int m = mtile * 16 + r + 8 * hi;
    xg[(size_t)m * GPAD + ntile * 16 + col] = acc[r] + bv;
  }
}

// ---------------- persistent LSTM recurrence ----------------
// 128 single-wave WGs; WG w owns (ntile = w>>1, mt = w&1): 16 hidden cols x 16 batch rows,
// all 4 gates. W_hh slice (4 x 1024 x 16 bf16 = 128KB) cached in LDS via async
// global->LDS copies. h double-buffered in global bf16; counter grid barrier per step.
__global__ void __launch_bounds__(32) lstm_kernel(const float* __restrict__ xg,
                                                  const unsigned short* __restrict__ Bhh,
                                                  unsigned short* __restrict__ h_buf,
                                                  float* __restrict__ hs,
                                                  unsigned* __restrict__ bar) {
  extern __shared__ unsigned short sB[];   // [4 gates][32 kb][32 lanes][16 bf16] = 128KB
  const int wg = blockIdx.x;               // 0..127
  const int ntile = wg >> 1;               // 0..63
  const int mt = wg & 1;                   // batch tile
  const int lane = threadIdx.x & 31;
  const int col = lane & 15, hi = lane >> 4;
  const unsigned nWG = gridDim.x;

  // ---- one-time fill of LDS weight cache with CDNA5 async global->LDS copies ----
  for (int fr = 0; fr < 128; ++fr) {       // fr = g*32 + kb
    int g = fr >> 5, kb = fr & 31;
    const unsigned short* src = Bhh + (size_t)(kb * 32 + lane) * GPAD + g * HPAD + ntile * 16;
    unsigned lds_off = (unsigned)((fr * 32 + lane) * 32);   // byte offset in LDS
    asm volatile("global_load_async_to_lds_b128 %0, %1, off"
                 :: "v"(lds_off), "v"(src) : "memory");
    asm volatile("global_load_async_to_lds_b128 %0, %1, off offset:16"
                 :: "v"(lds_off), "v"(src) : "memory");
  }
  asm volatile("s_wait_asynccnt 0" ::: "memory");

  v8f cc = vzero8();                       // persistent cell-state fragment

  for (int t = 0; t < SEQ; ++t) {
    const unsigned short* h_rd = h_buf + (size_t)(t & 1) * BATCH * HPAD;
    unsigned short* h_wr = h_buf + (size_t)((t + 1) & 1) * BATCH * HPAD;

    // prefetch next step's xg rows for this tile (global_prefetch_b8)
    if (t + 1 < SEQ) {
      const float* pf = xg + ((size_t)((t + 1) * BATCH) + mt * 16 + col) * GPAD
                           + (size_t)(hi * 2) * HPAD + ntile * 16;
      __builtin_prefetch(pf, 0, 1);
      __builtin_prefetch(pf + HPAD, 0, 1);
    }

    v8f acc[4];
#pragma unroll
    for (int g = 0; g < 4; ++g) acc[g] = vzero8();

    for (int kb = 0; kb < HPAD / 32; ++kb) {
      v16bf a = load_a(h_rd + (size_t)(mt * 16) * HPAD + kb * 32, HPAD);
#pragma unroll
      for (int g = 0; g < 4; ++g) {
        v16bf b = load_b_lds(sB + ((size_t)(g * 32 + kb) * 32) * 16);
        acc[g] = wmma_bf16(a, b, acc[g]);
      }
    }

#pragma unroll
    for (int r = 0; r < 8; ++r) {
      int b = mt * 16 + r + 8 * hi;                       // batch index 0..31
      size_t base = (size_t)(t * BATCH + b) * GPAD + ntile * 16 + col;
      float gi = acc[0][r] + xg[base + 0 * HPAD];
      float gf = acc[1][r] + xg[base + 1 * HPAD];
      float gg = acc[2][r] + xg[base + 2 * HPAD];
      float go = acc[3][r] + xg[base + 3 * HPAD];
      float si = 1.f / (1.f + __expf(-gi));
      float sf = 1.f / (1.f + __expf(-gf));
      float so = 1.f / (1.f + __expf(-go));
      float cn = sf * cc[r] + si * tanhf(gg);
      cc[r] = cn;
      float h = so * tanhf(cn);
      hs[(size_t)(t * BATCH + b) * HPAD + ntile * 16 + col] = h;
      h_wr[(size_t)b * HPAD + ntile * 16 + col] = f32_to_bf16_bits(h);
    }

    // grid-wide barrier between timesteps
    __threadfence();
    if (lane == 0) atomicAdd(bar, 1u);
    const unsigned target = (unsigned)(t + 1) * nWG;
    while (__hip_atomic_load(bar, __ATOMIC_ACQUIRE, __HIP_MEMORY_SCOPE_AGENT) < target) {}
  }
}

// ---------------- tag_space = hs @ lin_w^T + lin_b ----------------
__global__ void __launch_bounds__(288) linear_kernel(const float* __restrict__ hs,
                                                     const float* __restrict__ lin_w,
                                                     const float* __restrict__ lin_b,
                                                     float* __restrict__ ts) {
  int t = blockIdx.x;
  int tid = threadIdx.x;                 // 0..287
  int b = tid / TAGS, k = tid % TAGS;
  const float* hrow = hs + (size_t)(t * BATCH + b) * HPAD;
  const float* wrow = lin_w + (size_t)k * HIDDEN;
  float s = lin_b[k];
  for (int h = 0; h < HIDDEN; ++h) s = fmaf(hrow[h], wrow[h], s);
  ts[(size_t)(t * BATCH + b) * TAGS + k] = s;
}

// logsumexp over batch axis (reference's log_softmax axis=1)
__global__ void lseb_kernel(const float* __restrict__ ts, float* __restrict__ lseb) {
  int t = blockIdx.x;
  int k = threadIdx.x;
  if (k >= TAGS) return;
  float m = -1e30f;
  for (int b = 0; b < BATCH; ++b) m = fmaxf(m, ts[(size_t)(t * BATCH + b) * TAGS + k]);
  float s = 0.f;
  for (int b = 0; b < BATCH; ++b) s += __expf(ts[(size_t)(t * BATCH + b) * TAGS + k] - m);
  lseb[t * TAGS + k] = m + __logf(s);
}

// ---------------- CRF forward + numerator, single block ----------------
__global__ void __launch_bounds__(288) crf_kernel(const float* __restrict__ ts,
                                                  const float* __restrict__ lseb,
                                                  const int* __restrict__ y,
                                                  const float* __restrict__ start_t,
                                                  const float* __restrict__ end_t,
                                                  const float* __restrict__ trans,
                                                  float* __restrict__ out) {
  __shared__ float alpha[2][BATCH][TAGS];
  __shared__ float red[BATCH];
  int tid = threadIdx.x;
  int b = tid / TAGS, k = tid % TAGS;

  alpha[0][b][k] = start_t[k] + (ts[(size_t)b * TAGS + k] - lseb[k]);
  __syncthreads();
  int cur = 0;
  for (int t = 1; t < SEQ; ++t) {
    float v[TAGS], m = -1e30f;
#pragma unroll
    for (int j = 0; j < TAGS; ++j) { v[j] = alpha[cur][b][j] + trans[j * TAGS + k]; m = fmaxf(m, v[j]); }
    float s = 0.f;
#pragma unroll
    for (int j = 0; j < TAGS; ++j) s += __expf(v[j] - m);
    float em = ts[(size_t)(t * BATCH + b) * TAGS + k] - lseb[t * TAGS + k];
    alpha[cur ^ 1][b][k] = m + __logf(s) + em;
    cur ^= 1;
    __syncthreads();
  }
  if (k == 0) {
    float v[TAGS], m = -1e30f;
#pragma unroll
    for (int j = 0; j < TAGS; ++j) { v[j] = alpha[cur][b][j] + end_t[j]; m = fmaxf(m, v[j]); }
    float s = 0.f;
#pragma unroll
    for (int j = 0; j < TAGS; ++j) s += __expf(v[j] - m);
    float logz = m + __logf(s);

    float num = start_t[y[b]] + end_t[y[(SEQ - 1) * BATCH + b]];
    int yprev = y[b];
    for (int t = 0; t < SEQ; ++t) {
      int yt = y[t * BATCH + b];
      num += ts[(size_t)(t * BATCH + b) * TAGS + yt] - lseb[t * TAGS + yt];
      if (t > 0) num += trans[yprev * TAGS + yt];
      yprev = yt;
    }
    red[b] = num - logz;
  }
  __syncthreads();
  if (tid == 0) {
    float s = 0.f;
    for (int bb = 0; bb < BATCH; ++bb) s += red[bb];
    out[0] = s;
  }
}

// ---------------- host launcher ----------------
extern "C" void kernel_launch(void* const* d_in, const int* in_sizes, int n_in,
                              void* d_out, int out_size, void* d_ws, size_t ws_size,
                              hipStream_t stream) {
  (void)in_sizes; (void)n_in; (void)out_size; (void)ws_size;
  const int*   x        = (const int*)d_in[0];
  const int*   y        = (const int*)d_in[1];
  const float* embed_w  = (const float*)d_in[2];
  const float* W_ih     = (const float*)d_in[3];
  const float* W_hh     = (const float*)d_in[4];
  const float* b_ih     = (const float*)d_in[5];
  const float* b_hh     = (const float*)d_in[6];
  const float* lin_w    = (const float*)d_in[7];
  const float* lin_b    = (const float*)d_in[8];
  const float* start_t  = (const float*)d_in[9];
  const float* end_t    = (const float*)d_in[10];
  const float* trans    = (const float*)d_in[11];
  float* out = (float*)d_out;

  char* ws = (char*)d_ws;
  size_t off = 0;
  auto take = [&](size_t bytes) -> void* {
    off = (off + 255) & ~(size_t)255;
    void* p = ws + off;
    off += bytes;
    return p;
  };
  unsigned*       bar   = (unsigned*)      take(256);
  float*          bias  = (float*)         take((size_t)GPAD * 4);
  unsigned short* h_buf = (unsigned short*)take((size_t)2 * BATCH * HPAD * 2);
  unsigned short* Bih   = (unsigned short*)take((size_t)KPAD * GPAD * 2);
  unsigned short* Bhh   = (unsigned short*)take((size_t)HPAD * GPAD * 2);
  unsigned short* emb   = (unsigned short*)take((size_t)ROWS * KPAD * 2);
  float*          xg    = (float*)         take((size_t)ROWS * GPAD * 4);
  float*          hs    = (float*)         take((size_t)ROWS * HPAD * 4);
  float*          tsbuf = (float*)         take((size_t)ROWS * TAGS * 4);
  float*          lseb  = (float*)         take((size_t)SEQ * TAGS * 4);

  init_misc<<<dim3(((size_t)2 * BATCH * HPAD + 255) / 256), dim3(256), 0, stream>>>(b_ih, b_hh, bias, h_buf, bar);
  conv_wih<<<dim3(((size_t)KPAD * GPAD + 255) / 256), dim3(256), 0, stream>>>(W_ih, Bih);
  conv_whh<<<dim3(((size_t)HPAD * GPAD + 255) / 256), dim3(256), 0, stream>>>(W_hh, Bhh);
  embed_gather<<<dim3(((size_t)ROWS * KPAD + 255) / 256), dim3(256), 0, stream>>>(x, embed_w, emb);

  xg_gemm<<<dim3(64, ROWS / 16), dim3(128), 0, stream>>>(emb, Bih, bias, xg);

  // 128 WGs, 1 wave each, 128KB dynamic LDS for the W_hh slice cache
  lstm_kernel<<<dim3(128), dim3(32), 131072, stream>>>(xg, Bhh, h_buf, hs, bar);

  linear_kernel<<<dim3(SEQ), dim3(BATCH * TAGS), 0, stream>>>(hs, lin_w, lin_b, tsbuf);
  lseb_kernel<<<dim3(SEQ), dim3(32), 0, stream>>>(tsbuf, lseb);
  crf_kernel<<<dim3(1), dim3(BATCH * TAGS), 0, stream>>>(tsbuf, lseb, y, start_t, end_t, trans, out);
}
